// BidirectionalMamba_2731599200636
// MI455X (gfx1250) — matrix operand order repaired and checked by reading, not compiled, not run on hardware
//
#include <hip/hip_runtime.h>

// ---------------------------------------------------------------------------
// Bidirectional Mamba for MI455X (gfx1250)
//   * all GEMMs: v_wmma_f32_16x16x32_bf16, 128x64 block tiles, 8 waves
//   * GEMM tile staging: Tensor Data Mover (tensor_load_to_lds) double-buffered
//   * selective scan: 3-phase chunked parallel scan (serial depth 2048 -> 520)
// ---------------------------------------------------------------------------
#define BATCH   8
#define SEQ     2048
#define DMODEL  512
#define DINNER  1024
#define DSTATE  16
#define DCONV   4
#define DTRANK  32
#define BL      (BATCH * SEQ)      // 16384 rows
#define CH      256                // scan chunk length
#define NCH     (SEQ / CH)         // 8 chunks

typedef __attribute__((ext_vector_type(16))) __bf16 v16bf;
typedef __attribute__((ext_vector_type(8)))  float  v8f;
typedef __attribute__((ext_vector_type(4))) unsigned int tdm_u32x4;
typedef __attribute__((ext_vector_type(8))) int          tdm_i32x8;
typedef __attribute__((ext_vector_type(4))) int          tdm_i32x4;

#if __has_builtin(__builtin_amdgcn_tensor_load_to_lds)
#define HAVE_TDM 1
#else
#define HAVE_TDM 0
#endif

#if HAVE_TDM
#if __has_builtin(__builtin_amdgcn_s_wait_tensorcnt)
#define TDM_WAIT(n) __builtin_amdgcn_s_wait_tensorcnt((short)(n))
#else
#define TDM_WAIT(n) asm volatile("s_wait_tensorcnt %0" ::"i"(n))
#endif
#endif

union FragBF { unsigned short u[16]; v16bf v; };

static __device__ __forceinline__ unsigned short f32_to_bf16_rne(float f) {
    unsigned int u = __float_as_uint(f);
    u += 0x7FFFu + ((u >> 16) & 1u);     // round-to-nearest-even
    return (unsigned short)(u >> 16);
}

// ------------------------- fp32 -> bf16 conversion -------------------------
__global__ void cvt_bf16_kernel(const float* __restrict__ src,
                                unsigned short* __restrict__ dst, int n) {
    int i = blockIdx.x * blockDim.x + threadIdx.x;
    if (i < n) dst[i] = f32_to_bf16_rne(src[i]);
}

// ------------------------- WMMA bf16 GEMM: C = A @ B^T ---------------------
#define GBM 128
#define GBN 64
#define GBK 32
#define GPAD 8

static __device__ __forceinline__ void storeC(float* C, long idx, float v) { C[idx] = v; }
static __device__ __forceinline__ void storeC(unsigned short* C, long idx, float v) {
    C[idx] = f32_to_bf16_rne(v);
}

#if HAVE_TDM
// TDM 2-D tile load: rows x GBK bf16 tile from row-major [*,K] matrix into LDS,
// with pad_interval=16 DWORDs (one 64B row) / pad_amount=4 DWORDs (16B) so the
// LDS image has row stride (GBK+GPAD)*2 = 80B.
static __device__ __forceinline__ void tdm_load_tile_bf16(
    const unsigned short* gsrc, unsigned int lds_byte, int K, int rows)
{
    unsigned long long ga = (unsigned long long)gsrc;
    tdm_u32x4 g0;
    g0[0] = 1u;                                          // count=1 (valid)
    g0[1] = lds_byte;                                    // lds_addr
    g0[2] = (unsigned int)ga;                            // global_addr[31:0]
    g0[3] = (unsigned int)((ga >> 32) & 0x01FFFFFFu) | 0x80000000u; // [56:32]|type=2
    tdm_i32x8 g1;
    g1[0] = (int)((1u << 16)       // data_size = 2 bytes
                | (1u << 20)       // pad_enable
                | (3u << 22)       // pad_interval: 16 DWORDs (= one 64B tile row)
                | (3u << 25));     // pad_amount: 4 DWORDs (16B)
    g1[1] = (int)(((unsigned)K & 0xFFFFu) << 16);                 // tensor_dim0 lo16
    g1[2] = (int)((((unsigned)K >> 16) & 0xFFFFu)
                | (((unsigned)rows & 0xFFFFu) << 16));            // dim0 hi | dim1 lo
    g1[3] = (int)((unsigned)GBK << 16);                           // tile_dim0 = 32
    g1[4] = (int)((unsigned)rows & 0xFFFFu);                      // tile_dim1 = rows
    g1[5] = (int)(unsigned)K;                                     // dim0_stride lo32
    g1[6] = 0; g1[7] = 0;
    tdm_i32x4 gz4 = {0, 0, 0, 0};
    tdm_i32x8 gz8 = {0, 0, 0, 0, 0, 0, 0, 0};
    // 6-arg form (clang-23 / therock-10.0 headers): g0, g1, g2, g3, g4, cpol
    __builtin_amdgcn_tensor_load_to_lds(g0, g1, gz4, gz4, gz8, 0);
}
#endif

static __device__ __forceinline__ void wmma_step(
    const unsigned short (*As)[GBK + GPAD],
    const unsigned short (*Bs)[GBK + GPAD],
    int wm, int wn, int half, int l16, v8f acc[2][2])
{
    FragBF af[2], bfr[2];
    #pragma unroll
    for (int tm = 0; tm < 2; ++tm) {
        const int m = wm * 32 + tm * 16 + l16;
        #pragma unroll
        for (int j = 0; j < 16; ++j) {
            // A 16x32 bf16: half selects K 0..15 / 16..31
            const int k = half * 16 + j;
            af[tm].u[j] = As[m][k];
        }
    }
    #pragma unroll
    for (int tn = 0; tn < 2; ++tn) {
        const int n = wn * 32 + tn * 16 + l16;
        #pragma unroll
        for (int j = 0; j < 16; ++j)   // B 32x16: half selects K 0..15 / 16..31
            bfr[tn].u[j] = Bs[n][half * 16 + j];
    }
    #pragma unroll
    for (int tm = 0; tm < 2; ++tm)
        #pragma unroll
        for (int tn = 0; tn < 2; ++tn)
            acc[tm][tn] = __builtin_amdgcn_wmma_f32_16x16x32_bf16(
                false, af[tm].v, false, bfr[tn].v,
                (short)0, acc[tm][tn], false, false);
}

template <typename OutT>
__global__ __launch_bounds__(256) void gemm_bf16_nt(
    const unsigned short* __restrict__ A,   // [M,K] bf16
    const unsigned short* __restrict__ B,   // [N,K] bf16
    OutT* __restrict__ C, int M, int N, int K, int ldc)
{
#if HAVE_TDM
    __shared__ unsigned short Asm[2][GBM][GBK + GPAD];
    __shared__ unsigned short Bsm[2][GBN][GBK + GPAD];
#else
    __shared__ unsigned short Asm[1][GBM][GBK + GPAD];
    __shared__ unsigned short Bsm[1][GBN][GBK + GPAD];
#endif
    const int tid  = threadIdx.x;
    const int lane = tid & 31;
    const int wid  = tid >> 5;
    const int wm   = wid & 3;         // 4 waves along M (32 rows each)
    const int wn   = wid >> 2;        // 2 waves along N (32 cols each)
    const int half = lane >> 4;
    const int l16  = lane & 15;
    const long rowBase = (long)blockIdx.y * GBM;
    const long colBase = (long)blockIdx.x * GBN;

    v8f acc[2][2] = {{{}, {}}, {{}, {}}};
    const int nk = K / GBK;

#if HAVE_TDM
    if (wid == 0) {   // wave 0 drives the Tensor Data Mover
        tdm_load_tile_bf16(A + rowBase * (long)K,
                           (unsigned int)(unsigned long long)&Asm[0][0][0], K, GBM);
        tdm_load_tile_bf16(B + colBase * (long)K,
                           (unsigned int)(unsigned long long)&Bsm[0][0][0], K, GBN);
    }
    for (int i = 0; i < nk; ++i) {
        if (wid == 0) {
            if (i + 1 < nk) {
                // issue next tile pair into the other buffer, then wait until
                // only those 2 TDM ops remain outstanding (current pair done)
                tdm_load_tile_bf16(A + rowBase * (long)K + (long)(i + 1) * GBK,
                    (unsigned int)(unsigned long long)&Asm[(i + 1) & 1][0][0], K, GBM);
                tdm_load_tile_bf16(B + colBase * (long)K + (long)(i + 1) * GBK,
                    (unsigned int)(unsigned long long)&Bsm[(i + 1) & 1][0][0], K, GBN);
                TDM_WAIT(2);
            } else {
                TDM_WAIT(0);
            }
        }
        __syncthreads();
        wmma_step(Asm[i & 1], Bsm[i & 1], wm, wn, half, l16, acc);
        __syncthreads();
    }
#else
    for (int kb = 0; kb < K; kb += GBK) {
        for (int c = tid; c < (GBM * GBK / 8); c += 256) {
            int r = c >> 2, cg = (c & 3) * 8;
            *(uint4*)(&Asm[0][r][cg]) =
                *(const uint4*)(A + (rowBase + r) * (long)K + kb + cg);
        }
        for (int c = tid; c < (GBN * GBK / 8); c += 256) {
            int r = c >> 2, cg = (c & 3) * 8;
            *(uint4*)(&Bsm[0][r][cg]) =
                *(const uint4*)(B + (colBase + r) * (long)K + kb + cg);
        }
        __syncthreads();
        wmma_step(Asm[0], Bsm[0], wm, wn, half, l16, acc);
        __syncthreads();
    }
#endif

    // C/D layout: vgpr i -> m = i + 8*half, n = lane%16
    #pragma unroll
    for (int tm = 0; tm < 2; ++tm)
        #pragma unroll
        for (int tn = 0; tn < 2; ++tn)
            #pragma unroll
            for (int i = 0; i < 8; ++i) {
                const long m = rowBase + wm * 32 + tm * 16 + i + 8 * half;
                const long n = colBase + wn * 32 + tn * 16 + l16;
                storeC(C, m * (long)ldc + n, acc[tm][tn][i]);
            }
}

// --------------- causal / anti-causal depthwise conv + SiLU ----------------
__global__ void conv_silu_kernel(const float* __restrict__ xz,
                                 const float* __restrict__ cw,
                                 const float* __restrict__ cb,
                                 float* __restrict__ xc,
                                 unsigned short* __restrict__ xc_bf,
                                 int dir)
{
    int idx = blockIdx.x * blockDim.x + threadIdx.x;
    if (idx >= BL * DINNER) return;
    const int c = idx & (DINNER - 1);
    const int l = (idx / DINNER) % SEQ;
    const int b = idx / (DINNER * SEQ);

    float acc = cb[c];
    #pragma unroll
    for (int k = 0; k < DCONV; ++k) {
        const int pos = dir ? (l + (DCONV - 1) - k) : (l - (DCONV - 1) + k);
        if (pos >= 0 && pos < SEQ)
            acc += cw[c * DCONV + k] * xz[((long)(b * SEQ + pos)) * (2 * DINNER) + c];
    }
    const float s = acc / (1.0f + __expf(-acc));   // SiLU
    xc[idx]    = s;
    xc_bf[idx] = f32_to_bf16_rne(s);
}

// --------------- extract dt columns (first 32 of dbl) as bf16 --------------
__global__ void extract_dt_kernel(const float* __restrict__ dbl,
                                  unsigned short* __restrict__ dtr, int n) {
    int i = blockIdx.x * blockDim.x + threadIdx.x;
    if (i >= n) return;
    dtr[i] = f32_to_bf16_rne(dbl[(long)(i >> 5) * 64 + (i & 31)]);
}

// ----------------------- softplus(dt_proj + bias) --------------------------
__global__ void softplus_bias_kernel(float* __restrict__ dt,
                                     const float* __restrict__ bias, int n) {
    int i = blockIdx.x * blockDim.x + threadIdx.x;
    if (i >= n) return;
    const float v = dt[i] + bias[i & (DINNER - 1)];
    dt[i] = (v > 20.0f) ? v : log1pf(__expf(v));
}

// ------------------- chunked selective scan: phase A -----------------------
// Local scan of each chunk from h=0; emits chunk-final state + sum(dt).
// grid (DINNER/256, BATCH, NCH)
__global__ __launch_bounds__(256) void scan_phaseA(
    const float* __restrict__ xc, const float* __restrict__ dbl,
    const float* __restrict__ dt, const float* __restrict__ A_log,
    float* __restrict__ hend,     // [BATCH*NCH*DINNER*DSTATE]
    float* __restrict__ dtsum,    // [BATCH*NCH*DINNER]
    int dir)
{
    __shared__ float Bs[DSTATE];
    const int d = blockIdx.x * 256 + threadIdx.x;
    const int b = blockIdx.y;
    const int c = blockIdx.z;

    float A[DSTATE], h[DSTATE];
    #pragma unroll
    for (int s = 0; s < DSTATE; ++s) {
        A[s] = -__expf(A_log[d * DSTATE + s]);
        h[s] = 0.0f;
    }
    float dts = 0.0f;

    for (int t = c * CH; t < (c + 1) * CH; ++t) {
        const int  l   = dir ? (SEQ - 1 - t) : t;
        const long row = (long)b * SEQ + l;
        if (threadIdx.x < DSTATE)
            Bs[threadIdx.x] = dbl[row * 64 + 32 + threadIdx.x];
        __syncthreads();
        const float dtv = dt[row * DINNER + d];
        const float dtx = dtv * xc[row * DINNER + d];
        dts += dtv;
        #pragma unroll
        for (int s = 0; s < DSTATE; ++s)
            h[s] = __expf(dtv * A[s]) * h[s] + dtx * Bs[s];
        __syncthreads();
    }
    const long base = ((long)(b * NCH + c) * DINNER + d);
    #pragma unroll
    for (int s = 0; s < DSTATE; ++s) hend[base * DSTATE + s] = h[s];
    dtsum[base] = dts;
}

// ------------------- chunked selective scan: phase B -----------------------
// Carry scan across chunks; total chunk decay is exactly exp(A * sum(dt)).
// In-place: hend[c] (local final state) is replaced by h_in[c] (incoming state).
__global__ void scan_phaseB(const float* __restrict__ A_log,
                            const float* __restrict__ dtsum,
                            float* __restrict__ hcarry)
{
    int idx = blockIdx.x * blockDim.x + threadIdx.x;
    if (idx >= BATCH * DINNER * DSTATE) return;
    const int s = idx & (DSTATE - 1);
    const int d = (idx >> 4) & (DINNER - 1);
    const int b = idx >> 14;
    const float As = -__expf(A_log[d * DSTATE + s]);
    float H = 0.0f;
    for (int c = 0; c < NCH; ++c) {
        const long base = ((long)(b * NCH + c) * DINNER + d);
        const float hl = hcarry[base * DSTATE + s];
        const float E  = __expf(As * dtsum[base]);
        hcarry[base * DSTATE + s] = H;   // incoming state for chunk c
        H = E * H + hl;
    }
}

// ------------------- chunked selective scan: phase C -----------------------
// Re-run each chunk seeded with its incoming state; emit gated bf16 output.
__global__ __launch_bounds__(256) void scan_phaseC(
    const float* __restrict__ xz, const float* __restrict__ xc,
    const float* __restrict__ dbl, const float* __restrict__ dt,
    const float* __restrict__ A_log, const float* __restrict__ Dv,
    const float* __restrict__ hin,
    unsigned short* __restrict__ y_bf, int dir)
{
    __shared__ float Bs[DSTATE], Cs[DSTATE];
    const int d = blockIdx.x * 256 + threadIdx.x;
    const int b = blockIdx.y;
    const int c = blockIdx.z;

    float A[DSTATE], h[DSTATE];
    const long base = ((long)(b * NCH + c) * DINNER + d);
    #pragma unroll
    for (int s = 0; s < DSTATE; ++s) {
        A[s] = -__expf(A_log[d * DSTATE + s]);
        h[s] = hin[base * DSTATE + s];
    }
    const float Dd = Dv[d];

    for (int t = c * CH; t < (c + 1) * CH; ++t) {
        const int  l   = dir ? (SEQ - 1 - t) : t;
        const long row = (long)b * SEQ + l;
        if (threadIdx.x < 32) {
            const int   s = threadIdx.x;
            const float v = dbl[row * 64 + 32 + s];
            if (s < DSTATE) Bs[s] = v; else Cs[s - DSTATE] = v;
        }
        __syncthreads();
        const float dtv = dt[row * DINNER + d];
        const float xv  = xc[row * DINNER + d];
        const float zv  = xz[row * (2 * DINNER) + DINNER + d];
        const float dtx = dtv * xv;
        float y = 0.0f;
        #pragma unroll
        for (int s = 0; s < DSTATE; ++s) {
            h[s] = __expf(dtv * A[s]) * h[s] + dtx * Bs[s];
            y += h[s] * Cs[s];
        }
        y = (y + xv * Dd) * (zv / (1.0f + __expf(-zv)));
        y_bf[row * DINNER + d] = f32_to_bf16_rne(y);
        __syncthreads();
    }
}

// ---------------------------------------------------------------------------
extern "C" void kernel_launch(void* const* d_in, const int* in_sizes, int n_in,
                              void* d_out, int out_size, void* d_ws, size_t ws_size,
                              hipStream_t stream)
{
    (void)in_sizes; (void)n_in; (void)out_size; (void)ws_size;
    const float* x      = (const float*)d_in[0];
    const float* fuse_w = (const float*)d_in[19];

    char* ws = (char*)d_ws;
    size_t off = 0;
    auto alloc = [&](size_t bytes) -> void* {
        void* p = ws + off;
        off = (off + bytes + 255) & ~(size_t)255;
        return p;
    };

    unsigned short* x_bf    = (unsigned short*)alloc((size_t)BL * DMODEL * 2);
    unsigned short* w_in_bf = (unsigned short*)alloc((size_t)2 * DINNER * DMODEL * 2);
    unsigned short* w_xp_bf = (unsigned short*)alloc((size_t)(DTRANK + 2 * DSTATE) * DINNER * 2);
    unsigned short* w_dt_bf = (unsigned short*)alloc((size_t)DINNER * DTRANK * 2);
    unsigned short* w_ot_bf = (unsigned short*)alloc((size_t)DMODEL * DINNER * 2);
    unsigned short* w_fu_bf = (unsigned short*)alloc((size_t)DMODEL * (2 * DMODEL) * 2);
    float*          xz_f    = (float*)alloc((size_t)BL * 2 * DINNER * 4);
    float*          xc_f    = (float*)alloc((size_t)BL * DINNER * 4);
    unsigned short* xc_bf   = (unsigned short*)alloc((size_t)BL * DINNER * 2);
    float*          dbl_f   = (float*)alloc((size_t)BL * 64 * 4);
    unsigned short* dtr_bf  = (unsigned short*)alloc((size_t)BL * DTRANK * 2);
    float*          dt_f    = (float*)alloc((size_t)BL * DINNER * 4);
    unsigned short* y_bf    = (unsigned short*)alloc((size_t)BL * DINNER * 2);
    unsigned short* cat_bf  = (unsigned short*)alloc((size_t)BL * (2 * DMODEL) * 2);
    float*          hcarry  = (float*)alloc((size_t)BATCH * NCH * DINNER * DSTATE * 4);
    float*          dtsum   = (float*)alloc((size_t)BATCH * NCH * DINNER * 4);

    auto cvt = [&](const float* s, unsigned short* dte, int n) {
        cvt_bf16_kernel<<<(n + 255) / 256, 256, 0, stream>>>(s, dte, n);
    };

    cvt(x, x_bf, BL * DMODEL);
    cvt(fuse_w, w_fu_bf, DMODEL * 2 * DMODEL);

    for (int dir = 0; dir < 2; ++dir) {
        const int base = 1 + dir * 9;
        const float* in_proj_w = (const float*)d_in[base + 0];
        const float* conv_w    = (const float*)d_in[base + 1];
        const float* conv_b    = (const float*)d_in[base + 2];
        const float* x_proj_w  = (const float*)d_in[base + 3];
        const float* dt_proj_w = (const float*)d_in[base + 4];
        const float* dt_proj_b = (const float*)d_in[base + 5];
        const float* A_log     = (const float*)d_in[base + 6];
        const float* Dvec      = (const float*)d_in[base + 7];
        const float* out_w     = (const float*)d_in[base + 8];

        // in_proj: xz = x @ W^T   [16384,2048]
        cvt(in_proj_w, w_in_bf, 2 * DINNER * DMODEL);
        gemm_bf16_nt<float><<<dim3((2 * DINNER) / GBN, BL / GBM), 256, 0, stream>>>(
            x_bf, w_in_bf, xz_f, BL, 2 * DINNER, DMODEL, 2 * DINNER);

        // depthwise conv + SiLU (anti-causal when dir=1)
        conv_silu_kernel<<<(BL * DINNER + 255) / 256, 256, 0, stream>>>(
            xz_f, conv_w, conv_b, xc_f, xc_bf, dir);

        // x_proj: dbl = xc @ W^T   [16384,64]
        cvt(x_proj_w, w_xp_bf, (DTRANK + 2 * DSTATE) * DINNER);
        gemm_bf16_nt<float><<<dim3(64 / GBN, BL / GBM), 256, 0, stream>>>(
            xc_bf, w_xp_bf, dbl_f, BL, 64, DINNER, 64);

        // dt_proj + softplus
        extract_dt_kernel<<<(BL * DTRANK + 255) / 256, 256, 0, stream>>>(
            dbl_f, dtr_bf, BL * DTRANK);
        cvt(dt_proj_w, w_dt_bf, DINNER * DTRANK);
        gemm_bf16_nt<float><<<dim3(DINNER / GBN, BL / GBM), 256, 0, stream>>>(
            dtr_bf, w_dt_bf, dt_f, BL, DINNER, DTRANK, DINNER);
        softplus_bias_kernel<<<(BL * DINNER + 255) / 256, 256, 0, stream>>>(
            dt_f, dt_proj_b, BL * DINNER);

        // chunked selective scan
        scan_phaseA<<<dim3(DINNER / 256, BATCH, NCH), 256, 0, stream>>>(
            xc_f, dbl_f, dt_f, A_log, hcarry, dtsum, dir);
        scan_phaseB<<<(BATCH * DINNER * DSTATE + 255) / 256, 256, 0, stream>>>(
            A_log, dtsum, hcarry);
        scan_phaseC<<<dim3(DINNER / 256, BATCH, NCH), 256, 0, stream>>>(
            xz_f, xc_f, dbl_f, dt_f, A_log, Dvec, hcarry, y_bf, dir);

        // out projection straight into concat buffer half (bf16, ldc = 1024)
        cvt(out_w, w_ot_bf, DMODEL * DINNER);
        gemm_bf16_nt<unsigned short><<<dim3(DMODEL / GBN, BL / GBM), 256, 0, stream>>>(
            y_bf, w_ot_bf, cat_bf + dir * DMODEL, BL, DMODEL, DINNER, 2 * DMODEL);
    }

    // fuse: out = cat @ fuse_w^T   [16384,512] fp32
    gemm_bf16_nt<float><<<dim3(DMODEL / GBN, BL / GBM), 256, 0, stream>>>(
        cat_bf, w_fu_bf, (float*)d_out, BL, DMODEL, 2 * DMODEL, DMODEL);
}
